// egretblock_63745904607369
// MI455X (gfx1250) — matrix operand order repaired and checked by reading, not compiled, not run on hardware
//
#include <hip/hip_runtime.h>

// ---------------------------------------------------------------------------
// EGRET block: GATv2(N=50000, E=800000, D=128, H=1, ED=16) + BN + Conv1d(k=3)
// + BN on CDNA5 (gfx1250, wave32). Dense GEMMs use v_wmma_f32_16x16x32_f16.
// B operands are pre-packed to f16 in fragment lane order, staged in LDS per
// workgroup (ds_load_b128 in the inner loop); A operands are f16 row-major
// (two global_load_b128 per fragment).
// ---------------------------------------------------------------------------

#define DF   128   // node feature dim
#define EDF  16    // edge attr dim
#define BPK  (8 * 4 * 32 * 16)   // packed B fragment halfs per 128x128 matrix

typedef __attribute__((ext_vector_type(16))) _Float16 v16h;
typedef __attribute__((ext_vector_type(8)))  _Float16 v8h;
typedef __attribute__((ext_vector_type(8)))  float    v8f;

__device__ __forceinline__ float2 softmax2(const float* w) {
  float a = w[0], b = w[1];
  float mx = fmaxf(a, b);
  float ea = __expf(a - mx), eb = __expf(b - mx);
  float s = ea + eb;
  return make_float2(ea / s, eb / s);
}

__device__ __forceinline__ void atomicMaxFloat(float* addr, float v) {
  if (v >= 0.f) atomicMax((int*)addr, __float_as_int(v));
  else          atomicMin((unsigned int*)addr, __float_as_uint(v));
}

__device__ __forceinline__ v16h combine8(v8h lo, v8h hi) {
  v16h r;
#pragma unroll
  for (int i = 0; i < 8; ++i) { r[i] = lo[i]; r[8 + i] = hi[i]; }
  return r;
}

// A (16x32 f16, MxK): lane l+16*half holds row M=l; two contiguous 8-half
// K-groups => two 16-byte loads from a row-major f16 row.
__device__ __forceinline__ v16h load_a_f16(const _Float16* row, int kbase, int half) {
  v8h lo = *(const v8h*)(row + kbase + 16 * half);
  v8h hi = *(const v8h*)(row + kbase + 16 * half + 8);
  return combine8(lo, hi);
}

// Packed B fragment from LDS: 16 contiguous halfs per lane (2x ds_load_b128)
__device__ __forceinline__ v16h load_frag_lds(const _Float16* p) {
  v8h lo = *(const v8h*)(p);
  v8h hi = *(const v8h*)(p + 8);
  return combine8(lo, hi);
}

// ---------------------------------------------------------------------------
// K0: workspace init
// ---------------------------------------------------------------------------
__global__ void k_init(float* attn, float* lmax, float* denom, float* stats, int nN) {
  size_t stride = (size_t)gridDim.x * blockDim.x;
  size_t t0 = (size_t)blockIdx.x * blockDim.x + threadIdx.x;
  size_t nd = (size_t)nN * DF;
  for (size_t t = t0; t < nd; t += stride) attn[t] = 0.f;
  for (size_t t = t0; t < (size_t)nN; t += stride) {
    lmax[t] = -__builtin_huge_valf();
    denom[t] = 0.f;
  }
  for (size_t t = t0; t < 8 * DF; t += stride) stats[t] = 0.f;
}

// ---------------------------------------------------------------------------
// Prep: x (f32) -> x16 (f16, row-major)
// ---------------------------------------------------------------------------
__global__ void k_prep_x(const float* __restrict__ x, _Float16* __restrict__ x16,
                         size_t total) {
  size_t t = (size_t)blockIdx.x * blockDim.x + threadIdx.x;
  if (t < total) x16[t] = (_Float16)x[t];
}

// ---------------------------------------------------------------------------
// Prep: pack 128(K) x 128(N) row-major f32 weight into B-fragment order:
//   Bp[((nt*4+ks)*32+lane)*16+p] = W[K*128+col], K=ks*32+16*half+p, col=nt*16+l
// ---------------------------------------------------------------------------
__global__ void k_pack_b(const float* __restrict__ W, _Float16* __restrict__ Bp) {
  int t = blockIdx.x * blockDim.x + threadIdx.x;
  if (t >= BPK) return;
  int p = t & 15, lane = (t >> 4) & 31, ks = (t >> 9) & 3, nt = (t >> 11) & 7;
  int half = lane >> 4, l = lane & 15;
  int K = ks * 32 + 16 * half + p;
  int col = nt * 16 + l;
  Bp[t] = (_Float16)W[(size_t)K * DF + col];
}

// Pack conv_w[co][ci][tap] into per-tap B-fragment order (B[K=ci][N=co])
__global__ void k_pack_conv(const float* __restrict__ cw, _Float16* __restrict__ Bp) {
  int t = blockIdx.x * blockDim.x + threadIdx.x;
  if (t >= 3 * BPK) return;
  int p = t & 15, lane = (t >> 4) & 31, ks = (t >> 9) & 3, nt = (t >> 11) & 7;
  int tap = t >> 14;
  int half = lane >> 4, l = lane & 15;
  int K = ks * 32 + 16 * half + p;
  int col = nt * 16 + l;
  Bp[t] = (_Float16)cw[(size_t)col * (3 * DF) + K * 3 + tap];
}

// ---------------------------------------------------------------------------
// K1: x_l = x@W_l + b_l ; x_r = x@W_r + b_r
//     Packed B staged in LDS (64 KB dynamic); full-tile fast-path stores.
// ---------------------------------------------------------------------------
__global__ void __launch_bounds__(256)
k_node_gemm(const _Float16* __restrict__ x16,
            const _Float16* __restrict__ BpL, const _Float16* __restrict__ BpR,
            const float* __restrict__ bl, const float* __restrict__ br,
            float* __restrict__ xl, float* __restrict__ xr, int nN) {
  extern __shared__ _Float16 sB[];   // 2*BPK halfs = 64 KB
  // cooperative stage: 4096 16-byte chunks
  {
    const uint4* gL = (const uint4*)BpL;
    const uint4* gR = (const uint4*)BpR;
    uint4* s = (uint4*)sB;
    for (int i = threadIdx.x; i < 4096; i += 256)
      s[i] = (i < 2048) ? gL[i] : gR[i - 2048];
  }
  __syncthreads();

  int wave = (blockIdx.x * blockDim.x + threadIdx.x) >> 5;
  int lane = threadIdx.x & 31;
  int half = lane >> 4, l = lane & 15;
  int n0 = wave * 16;
  if (n0 >= nN) return;
  int node = n0 + l;
  const _Float16* row = x16 + (size_t)(node < nN ? node : nN - 1) * DF;
  bool full = (n0 + 16 <= nN);

  v16h a[4];
#pragma unroll
  for (int ks = 0; ks < 4; ++ks) a[ks] = load_a_f16(row, ks * 32, half);

  for (int which = 0; which < 2; ++which) {
    const _Float16* Bs  = sB + which * BPK;
    const float* bias   = which ? br : bl;
    float*       out    = which ? xr : xl;
#pragma unroll
    for (int nt = 0; nt < 8; ++nt) {
      int col = nt * 16 + l;
      v8f acc = {};
#pragma unroll
      for (int ks = 0; ks < 4; ++ks) {
        v16h b = load_frag_lds(Bs + (((nt * 4 + ks) * 32 + lane) << 4));
        acc = __builtin_amdgcn_wmma_f32_16x16x32_f16(false, a[ks], false, b,
                                                     (short)0, acc, false, false);
      }
      float bv = bias[col];
      float* op = out + (size_t)(n0 + 8 * half) * DF + col;
      if (full) {
#pragma unroll
        for (int r = 0; r < 8; ++r) op[(size_t)r * DF] = acc[r] + bv;
      } else {
#pragma unroll
        for (int r = 0; r < 8; ++r)
          if (n0 + r + 8 * half < nN) op[(size_t)r * DF] = acc[r] + bv;
      }
    }
  }
}

// ---------------------------------------------------------------------------
// K2: per-edge logits (fused e = edge_attr@W_e), leaky(0.2), att-dot,
//     wave-reduce, segment max via float atomic max. One wave per edge.
// ---------------------------------------------------------------------------
__global__ void k_edge_logits(const float* __restrict__ xl, const float* __restrict__ xr,
                              const int* __restrict__ srcI, const int* __restrict__ dstI,
                              const float* __restrict__ ea, const float* __restrict__ We,
                              const float* __restrict__ att,
                              float* __restrict__ logits, float* __restrict__ lmax,
                              int E_, int nN) {
  int wave = (blockIdx.x * blockDim.x + threadIdx.x) >> 5;
  int lane = threadIdx.x & 31;
  int total = E_ + nN;
  if (wave >= total) return;
  int s, d;
  if (wave < E_) { s = srcI[wave]; d = dstI[wave]; }
  else           { s = d = wave - E_; }

  int c0 = lane * 4;
  float4 xlv = *(const float4*)(xl + (size_t)s * DF + c0);
  float4 xrv = *(const float4*)(xr + (size_t)d * DF + c0);
  float4 ev = make_float4(0.f, 0.f, 0.f, 0.f);
  if (wave < E_) {
    const float* ear = ea + (size_t)wave * EDF;
#pragma unroll
    for (int k = 0; k < EDF; ++k) {
      float av = ear[k];
      const float* wr = We + k * DF + c0;
      ev.x += av * wr[0]; ev.y += av * wr[1];
      ev.z += av * wr[2]; ev.w += av * wr[3];
    }
  }
  float4 attv = *(const float4*)(att + c0);
  float m, logit = 0.f;
  m = xlv.x + xrv.x + ev.x; m = m > 0.f ? m : 0.2f * m; logit += attv.x * m;
  m = xlv.y + xrv.y + ev.y; m = m > 0.f ? m : 0.2f * m; logit += attv.y * m;
  m = xlv.z + xrv.z + ev.z; m = m > 0.f ? m : 0.2f * m; logit += attv.z * m;
  m = xlv.w + xrv.w + ev.w; m = m > 0.f ? m : 0.2f * m; logit += attv.w * m;
#pragma unroll
  for (int off = 16; off; off >>= 1) logit += __shfl_xor(logit, off, 32);
  if (lane == 0) {
    logits[wave] = logit;
    atomicMaxFloat(&lmax[d], logit);
  }
}

// ---------------------------------------------------------------------------
// K3: p = exp(logit - lmax[dst]); denom[dst] += p
// ---------------------------------------------------------------------------
__global__ void k_edge_exp(float* __restrict__ logits, const float* __restrict__ lmax,
                           const int* __restrict__ dstI, float* __restrict__ denom,
                           int E_, int nN) {
  int t = blockIdx.x * blockDim.x + threadIdx.x;
  if (t >= E_ + nN) return;
  int d = (t < E_) ? dstI[t] : (t - E_);
  float p = __expf(logits[t] - lmax[d]);
  logits[t] = p;
  atomicAdd(&denom[d], p);
}

// ---------------------------------------------------------------------------
// K4: attn[dst] += (p/denom[dst]) * x_l[src]   (one wave per edge, 4 f/lane)
// ---------------------------------------------------------------------------
__global__ void k_edge_scatter(const float* __restrict__ p, const float* __restrict__ denom,
                               const int* __restrict__ srcI, const int* __restrict__ dstI,
                               const float* __restrict__ xl, float* __restrict__ attn,
                               int E_, int nN) {
  int wave = (blockIdx.x * blockDim.x + threadIdx.x) >> 5;
  int lane = threadIdx.x & 31;
  if (wave >= E_ + nN) return;
  int s, d;
  if (wave < E_) { s = srcI[wave]; d = dstI[wave]; }
  else           { s = d = wave - E_; }
  float alpha = p[wave] / denom[d];
  int c0 = lane * 4;
  float4 xv = *(const float4*)(xl + (size_t)s * DF + c0);
  float* o = attn + (size_t)d * DF + c0;
  atomicAdd(o + 0, alpha * xv.x);
  atomicAdd(o + 1, alpha * xv.y);
  atomicAdd(o + 2, alpha * xv.z);
  atomicAdd(o + 3, alpha * xv.w);
}

// ---------------------------------------------------------------------------
// K5: z0 = w1[0]*x + w1[1]*(attn + bias_gat); accumulate BN1 channel stats.
// ---------------------------------------------------------------------------
__global__ void k_combine_bn1(const float* __restrict__ x, const float* __restrict__ attn,
                              const float* __restrict__ bias_gat,
                              const float* __restrict__ weight1,
                              float* __restrict__ z0, float* __restrict__ sum1,
                              float* __restrict__ sumsq1, int nN) {
  int c = threadIdx.x;  // blockDim.x == 128
  float2 w = softmax2(weight1);
  float bg = bias_gat[c];
  float s = 0.f, sq = 0.f;
  for (int n = blockIdx.x; n < nN; n += gridDim.x) {
    size_t i = (size_t)n * DF + c;
    float v = w.x * x[i] + w.y * (attn[i] + bg);
    z0[i] = v;
    s += v; sq += v * v;
  }
  atomicAdd(&sum1[c], s);
  atomicAdd(&sumsq1[c], sq);
}

// ---------------------------------------------------------------------------
// BN finalize: per-channel scale/shift (biased variance, eps=1e-5)
// ---------------------------------------------------------------------------
__global__ void k_bn_finalize(const float* __restrict__ sum, const float* __restrict__ sumsq,
                              const float* __restrict__ gamma, const float* __restrict__ beta,
                              float* __restrict__ scale, float* __restrict__ shift, int nN) {
  int c = threadIdx.x;  // 128
  float inv = 1.f / (float)nN;
  float mu = sum[c] * inv;
  float var = sumsq[c] * inv - mu * mu;
  float sc = gamma[c] * rsqrtf(var + 1e-5f);
  scale[c] = sc;
  shift[c] = beta[c] - mu * sc;
}

// ---------------------------------------------------------------------------
// Prep: z16 = f16(BN1(z0))  (conv A operand)
// ---------------------------------------------------------------------------
__global__ void k_prep_z0n(const float* __restrict__ z0, const float* __restrict__ scale,
                           const float* __restrict__ shift, _Float16* __restrict__ z16,
                           size_t total) {
  size_t t = (size_t)blockIdx.x * blockDim.x + threadIdx.x;
  if (t >= total) return;
  int c = (int)(t & (DF - 1));
  z16[t] = (_Float16)(z0[t] * scale[c] + shift[c]);
}

// ---------------------------------------------------------------------------
// K6: conv1d(k=3 over nodes) as 3-tap WMMA GEMM; conv weights staged in LDS
//     (96 KB dynamic). Epilogue: leaky(0.01), z2 = w2[0]*BN1(z0) + w2[1]*z1,
//     BN2 stats.
// ---------------------------------------------------------------------------
__global__ void __launch_bounds__(256)
k_conv_wmma(const _Float16* __restrict__ z16, const float* __restrict__ z0,
            const float* __restrict__ scale1, const float* __restrict__ shift1,
            const _Float16* __restrict__ BpC, const float* __restrict__ cb,
            const float* __restrict__ weight2,
            float* __restrict__ z2, float* __restrict__ sum2,
            float* __restrict__ sumsq2, int nN) {
  extern __shared__ _Float16 sB[];   // 3*BPK halfs = 96 KB
  {
    const uint4* g = (const uint4*)BpC;
    uint4* s = (uint4*)sB;
    for (int i = threadIdx.x; i < 6144; i += 256) s[i] = g[i];
  }
  __syncthreads();

  int wave = (blockIdx.x * blockDim.x + threadIdx.x) >> 5;
  int lane = threadIdx.x & 31;
  int half = lane >> 4, l = lane & 15;
  int n0 = wave * 16;
  if (n0 >= nN) return;
  float2 w = softmax2(weight2);
  bool full = (n0 + 16 <= nN);

  v16h a[3][4];
#pragma unroll
  for (int tap = 0; tap < 3; ++tap) {
    int ni = n0 + l + tap - 1;
    bool valid = (ni >= 0 && ni < nN);
    const _Float16* row = z16 + (size_t)(valid ? ni : 0) * DF;
#pragma unroll
    for (int ks = 0; ks < 4; ++ks) {
      if (valid) a[tap][ks] = load_a_f16(row, ks * 32, half);
      else       a[tap][ks] = (v16h){};
    }
  }

  for (int nt = 0; nt < 8; ++nt) {
    int col = nt * 16 + l;
    v8f acc = {};
#pragma unroll
    for (int tap = 0; tap < 3; ++tap) {
#pragma unroll
      for (int ks = 0; ks < 4; ++ks) {
        v16h b = load_frag_lds(sB + (tap << 14) + (((nt * 4 + ks) * 32 + lane) << 4));
        acc = __builtin_amdgcn_wmma_f32_16x16x32_f16(false, a[tap][ks], false, b,
                                                     (short)0, acc, false, false);
      }
    }
    float cbv = cb[col];
    float sc = scale1[col], sh = shift1[col];
    float s = 0.f, sq = 0.f;
    const float* zp = z0 + (size_t)(n0 + 8 * half) * DF + col;
    float*       op = z2 + (size_t)(n0 + 8 * half) * DF + col;
    if (full) {
#pragma unroll
      for (int r = 0; r < 8; ++r) {
        float z1v = acc[r] + cbv;
        z1v = z1v > 0.f ? z1v : 0.01f * z1v;
        float zn = zp[(size_t)r * DF] * sc + sh;
        float v = w.x * zn + w.y * z1v;
        op[(size_t)r * DF] = v;
        s += v; sq += v * v;
      }
    } else {
#pragma unroll
      for (int r = 0; r < 8; ++r) {
        if (n0 + r + 8 * half < nN) {
          float z1v = acc[r] + cbv;
          z1v = z1v > 0.f ? z1v : 0.01f * z1v;
          float zn = zp[(size_t)r * DF] * sc + sh;
          float v = w.x * zn + w.y * z1v;
          op[(size_t)r * DF] = v;
          s += v; sq += v * v;
        }
      }
    }
    atomicAdd(&sum2[col], s);
    atomicAdd(&sumsq2[col], sq);
  }
}

// ---------------------------------------------------------------------------
// K7: out = BN2(z2)
// ---------------------------------------------------------------------------
__global__ void k_apply_bn2(const float* __restrict__ z2, const float* __restrict__ scale2,
                            const float* __restrict__ shift2, float* __restrict__ out,
                            size_t total) {
  size_t t = (size_t)blockIdx.x * blockDim.x + threadIdx.x;
  if (t >= total) return;
  int c = (int)(t & (DF - 1));
  out[t] = z2[t] * scale2[c] + shift2[c];
}

// ---------------------------------------------------------------------------
extern "C" void kernel_launch(void* const* d_in, const int* in_sizes, int n_in,
                              void* d_out, int out_size, void* d_ws, size_t ws_size,
                              hipStream_t stream) {
  const float* x        = (const float*)d_in[0];
  const int*   ei       = (const int*)  d_in[1];
  const float* ea       = (const float*)d_in[2];
  const float* Wl       = (const float*)d_in[3];
  const float* bl       = (const float*)d_in[4];
  const float* Wr       = (const float*)d_in[5];
  const float* br       = (const float*)d_in[6];
  const float* We       = (const float*)d_in[7];
  const float* att      = (const float*)d_in[8];
  const float* bias_gat = (const float*)d_in[9];
  const float* weight1  = (const float*)d_in[10];
  const float* g1       = (const float*)d_in[11];
  const float* b1       = (const float*)d_in[12];
  const float* cw       = (const float*)d_in[13];
  const float* cb       = (const float*)d_in[14];
  const float* weight2  = (const float*)d_in[15];
  const float* g2       = (const float*)d_in[16];
  const float* b2       = (const float*)d_in[17];

  const int nN = in_sizes[0] / DF;   // 50000
  const int E_ = in_sizes[1] / 2;    // 800000
  const int* srcI = ei;
  const int* dstI = ei + E_;

  // ---- workspace layout -------------------------------------------------
  float* ws    = (float*)d_ws;
  size_t nd    = (size_t)nN * DF;
  float* xl    = ws;
  float* xr    = xl + nd;
  float* attn  = xr + nd;
  float* z0    = attn + nd;
  float* z2    = attn;                 // alias: attn dead after k_combine_bn1
  float* logit = z0 + nd;              // E_+nN
  float* lmax  = logit + (size_t)(E_ + nN);
  float* denom = lmax + nN;
  float* stats = denom + nN;           // 8*DF
  float* sum1 = stats,          *sumsq1 = stats + DF;
  float* scl1 = stats + 2 * DF, *shf1   = stats + 3 * DF;
  float* sum2 = stats + 4 * DF, *sumsq2 = stats + 5 * DF;
  float* scl2 = stats + 6 * DF, *shf2   = stats + 7 * DF;
  _Float16* h16  = (_Float16*)(stats + 8 * DF);
  _Float16* x16  = h16;                // nd halfs
  _Float16* z16  = x16 + nd;           // nd halfs
  _Float16* BpL  = z16 + nd;           // BPK halfs
  _Float16* BpR  = BpL + BPK;          // BPK halfs
  _Float16* BpC  = BpR + BPK;          // 3*BPK halfs

  const int total_edges = E_ + nN;

  // K0: init
  k_init<<<1024, 256, 0, stream>>>(attn, lmax, denom, stats, nN);

  // Prep: f16 operand packing
  k_prep_x<<<(int)((nd + 255) / 256), 256, 0, stream>>>(x, x16, nd);
  k_pack_b<<<(BPK + 255) / 256, 256, 0, stream>>>(Wl, BpL);
  k_pack_b<<<(BPK + 255) / 256, 256, 0, stream>>>(Wr, BpR);
  k_pack_conv<<<(3 * BPK + 255) / 256, 256, 0, stream>>>(cw, BpC);

  // K1: node GEMMs (one wave per 16 nodes; 64 KB LDS of packed B)
  {
    int waves = (nN + 15) / 16;
    int blocks = (waves * 32 + 255) / 256;
    size_t shmem = (size_t)2 * BPK * sizeof(_Float16);
    k_node_gemm<<<blocks, 256, shmem, stream>>>(x16, BpL, BpR, bl, br, xl, xr, nN);
  }

  // K2: edge logits + segment max
  {
    long long threads = (long long)total_edges * 32;
    int blocks = (int)((threads + 255) / 256);
    k_edge_logits<<<blocks, 256, 0, stream>>>(xl, xr, srcI, dstI, ea, We, att,
                                              logit, lmax, E_, nN);
  }

  // K3: exp + segment sum
  {
    int blocks = (total_edges + 255) / 256;
    k_edge_exp<<<blocks, 256, 0, stream>>>(logit, lmax, dstI, denom, E_, nN);
  }

  // K4: weighted scatter
  {
    long long threads = (long long)total_edges * 32;
    int blocks = (int)((threads + 255) / 256);
    k_edge_scatter<<<blocks, 256, 0, stream>>>(logit, denom, srcI, dstI, xl, attn, E_, nN);
  }

  // K5: combine + BN1 stats
  k_combine_bn1<<<1024, DF, 0, stream>>>(x, attn, bias_gat, weight1, z0, sum1, sumsq1, nN);

  // BN1 finalize
  k_bn_finalize<<<1, DF, 0, stream>>>(sum1, sumsq1, g1, b1, scl1, shf1, nN);

  // Prep conv A operand: z16 = f16(BN1(z0))
  k_prep_z0n<<<(int)((nd + 255) / 256), 256, 0, stream>>>(z0, scl1, shf1, z16, nd);

  // K6: conv1d WMMA (96 KB LDS of packed conv weights)
  {
    int waves = (nN + 15) / 16;
    int blocks = (waves * 32 + 255) / 256;
    size_t shmem = (size_t)3 * BPK * sizeof(_Float16);
    k_conv_wmma<<<blocks, 256, shmem, stream>>>(z16, z0, scl1, shf1, BpC, cb, weight2,
                                                z2, sum2, sumsq2, nN);
  }

  // BN2 finalize
  k_bn_finalize<<<1, DF, 0, stream>>>(sum2, sumsq2, g2, b2, scl2, shf2, nN);

  // K7: apply BN2 -> output
  {
    int blocks = (int)((nd + 255) / 256);
    k_apply_bn2<<<blocks, 256, 0, stream>>>(z2, scl2, shf2, (float*)d_out, nd);
  }
}
